// ExpBertSelfAttention_75144747811240
// MI455X (gfx1250) — matrix operand order repaired and checked by reading, not compiled, not run on hardware
//
#include <hip/hip_runtime.h>

typedef __bf16 bf16;
typedef __attribute__((ext_vector_type(16))) __bf16 v16bf;
typedef __attribute__((ext_vector_type(8)))  float  v8f;
typedef __attribute__((ext_vector_type(4)))  int    v4i;

union FragAB { v16bf v; uint4 q[2]; };

#define WMMA_BF16(a, b, c) \
  __builtin_amdgcn_wmma_f32_16x16x32_bf16(false, (a), false, (b), (short)0, (c), false, false)

static constexpr int Bd = 2, Sd = 2048, Dd = 1024, Hd = 16, HDd = 64;

// ---------------------------------------------------------------------------
// Async global->LDS copy (CDNA5 GLOBAL_LOAD_ASYNC_TO_LDS_B128), with fallback.
// ---------------------------------------------------------------------------
#if defined(__HIP_DEVICE_COMPILE__) && __has_builtin(__builtin_amdgcn_global_load_async_to_lds_b128)
#define USE_ASYNC_LDS 1
typedef __attribute__((address_space(1))) v4i* gp4;
typedef __attribute__((address_space(3))) v4i* lp4;
#else
#define USE_ASYNC_LDS 0
#endif

__device__ __forceinline__ void cp16_g2l(void* lds, const void* g) {
#if USE_ASYNC_LDS
  __builtin_amdgcn_global_load_async_to_lds_b128(
      (gp4)(uintptr_t)g, (lp4)(unsigned)(uintptr_t)lds, 0, 0);
#else
  *(uint4*)lds = *(const uint4*)g;
#endif
}
__device__ __forceinline__ void wait_async_le6() {
#if USE_ASYNC_LDS
  asm volatile("s_wait_asynccnt 0x6" ::: "memory");
#endif
}
__device__ __forceinline__ void wait_async_0() {
#if USE_ASYNC_LDS
  asm volatile("s_wait_asynccnt 0x0" ::: "memory");
#endif
}

// ---------------------------------------------------------------------------
// 16-lane-row max butterfly: v_permlane16_b32 (VALU) instead of ds_bpermute.
// ---------------------------------------------------------------------------
#if defined(__HIP_DEVICE_COMPILE__) && __has_builtin(__builtin_amdgcn_permlane16)
__device__ __forceinline__ float pl16(float v, unsigned s1, unsigned s2) {
  int iv = __builtin_bit_cast(int, v);
  int r  = __builtin_amdgcn_permlane16(iv, iv, s1, s2, false, false);
  return __builtin_bit_cast(float, r);
}
__device__ __forceinline__ float rowmax16(float v) {
  v = fmaxf(v, pl16(v, 0x67452301u, 0xEFCDAB89u));  // xor 1
  v = fmaxf(v, pl16(v, 0x54761032u, 0xDCFE98BAu));  // xor 2
  v = fmaxf(v, pl16(v, 0x32107654u, 0xBA98FEDCu));  // xor 4
  v = fmaxf(v, pl16(v, 0xFEDCBA98u, 0x76543210u));  // xor 8
  return v;
}
#else
__device__ __forceinline__ float rowmax16(float v) {
  for (int sh = 1; sh < 16; sh <<= 1) v = fmaxf(v, __shfl_xor(v, sh, 32));
  return v;
}
#endif

// ---------------------------------------------------------------------------
// fp32 -> bf16 (n4 = n/4)
// ---------------------------------------------------------------------------
__global__ __launch_bounds__(256) void cvt_f32_bf16(const float* __restrict__ in,
                                                    bf16* __restrict__ out, int n4) {
  int i = blockIdx.x * blockDim.x + threadIdx.x;
  if (i < n4) {
    float4 f = reinterpret_cast<const float4*>(in)[i];
    bf16* o = out + (size_t)i * 4;
    o[0] = (bf16)f.x; o[1] = (bf16)f.y; o[2] = (bf16)f.z; o[3] = (bf16)f.w;
  }
}

// ---------------------------------------------------------------------------
// fp32 [K,N] -> bf16 [N,K] transpose-convert (32x32 tiles)
// ---------------------------------------------------------------------------
__global__ __launch_bounds__(256) void cvt_transpose(const float* __restrict__ in,
                                                     bf16* __restrict__ out,
                                                     int K, int N) {
  __shared__ bf16 t[32][33];
  const int k0 = blockIdx.x * 32, n0 = blockIdx.y * 32;
  const int r = threadIdx.x >> 3;
  const int c = (threadIdx.x & 7) * 4;
#pragma unroll
  for (int j = 0; j < 4; ++j)
    t[r][c + j] = (bf16)in[(size_t)(k0 + r) * N + n0 + c + j];
  __syncthreads();
#pragma unroll
  for (int j = 0; j < 4; ++j)
    out[(size_t)(n0 + r) * K + k0 + c + j] = t[c + j][r];
}

// ---------------------------------------------------------------------------
// bf16 GEMM: C[M,N] = A[M,K] @ Bt[N,K]^T + bias.  WG 128x256, wave 64x64,
// K-step 32, double-buffered async LDS staging (48 KB of 320 KB WGP LDS).
// ---------------------------------------------------------------------------
#define GM_TM 128
#define GM_TN 256
#define GM_TK 32

template <bool OUT_BF16, int N, int K>
__global__ __launch_bounds__(256) void gemm_bf16(const bf16* __restrict__ A,
                                                 const bf16* __restrict__ Bt,
                                                 const float* __restrict__ bias,
                                                 void* __restrict__ outPtr, int M) {
  __shared__ __align__(16) bf16 lA [2][GM_TM][GM_TK];   // 2 x 8 KB
  __shared__ __align__(16) bf16 lBt[2][GM_TN][GM_TK];   // 2 x 16 KB

  const int tid  = threadIdx.x;
  const int lane = tid & 31;
  const int wave = tid >> 5;
  const int laneN    = lane & 15;
  const int laneHalf = lane >> 4;

  constexpr int nTilesN = N / GM_TN;
  const int m0 = (blockIdx.x / nTilesN) * GM_TM;
  const int n0 = (blockIdx.x % nTilesN) * GM_TN;
  const int mw = (wave >> 2) * 64;
  const int nw = (wave & 3) * 64;

  auto stage = [&](int k0c, int buf) {
#pragma unroll
    for (int it = 0; it < 2; ++it) {                    // A: two 16B chunks
      int c   = tid + it * 256;                         // 512 chunks
      int row = c >> 2;
      int col = (c & 3) * 8;
      cp16_g2l(&lA[buf][row][col], A + (size_t)(m0 + row) * K + k0c + col);
    }
#pragma unroll
    for (int it = 0; it < 4; ++it) {                    // B: four 16B chunks
      int c  = tid + it * 256;                          // 1024 chunks
      int n  = c >> 2;
      int kc = (c & 3) * 8;
      cp16_g2l(&lBt[buf][n][kc], Bt + (size_t)(n0 + n) * K + k0c + kc);
    }
  };

  v8f acc[4][4];
#pragma unroll
  for (int i = 0; i < 4; ++i)
#pragma unroll
    for (int j = 0; j < 4; ++j) acc[i][j] = (v8f){0,0,0,0,0,0,0,0};

  stage(0, 0);
  constexpr int nk = K / GM_TK;
  for (int kt = 0; kt < nk; ++kt) {
    const int buf = kt & 1;
    if (kt + 1 < nk) { stage((kt + 1) * GM_TK, buf ^ 1); wait_async_le6(); }
    else             { wait_async_0(); }
    __syncthreads();                                    // tile kt visible to all

    FragAB af[4];
    {
      const int ac = laneHalf * 8;                      // A frag layout
#pragma unroll
      for (int fm = 0; fm < 4; ++fm) {
        const int row = mw + fm * 16 + laneN;
        af[fm].q[0] = *(uint4*)&lA[buf][row][ac];
        af[fm].q[1] = *(uint4*)&lA[buf][row][16 + ac];
      }
    }
#pragma unroll
    for (int fn = 0; fn < 4; ++fn) {
      FragAB bfrag;                                     // B frag layout
      const int n  = nw + fn * 16 + laneN;
      const int bk = laneHalf * 8;
      bfrag.q[0] = *(uint4*)&lBt[buf][n][bk];
      bfrag.q[1] = *(uint4*)&lBt[buf][n][bk + 16];
#pragma unroll
      for (int fm = 0; fm < 4; ++fm)
        acc[fm][fn] = WMMA_BF16(af[fm].v, bfrag.v, acc[fm][fn]);
    }

    __syncthreads();                                    // readers done with buf
  }

  // --- epilogue: constant-folded immediate-offset stores
  const int nwl = n0 + nw + laneN;
  float bv[4];
#pragma unroll
  for (int fn = 0; fn < 4; ++fn) bv[fn] = bias[nwl + fn * 16];

#pragma unroll
  for (int fm = 0; fm < 4; ++fm) {
    const size_t rowBase = (size_t)(m0 + mw + fm * 16 + laneHalf * 8) * N + nwl;
    if constexpr (OUT_BF16) {
      bf16* p = (bf16*)outPtr + rowBase;
#pragma unroll
      for (int j = 0; j < 8; ++j)
#pragma unroll
        for (int fn = 0; fn < 4; ++fn)
          p[j * N + fn * 16] = (bf16)(acc[fm][fn][j] + bv[fn]);
    } else {
      float* p = (float*)outPtr + rowBase;
#pragma unroll
      for (int j = 0; j < 8; ++j)
#pragma unroll
        for (int fn = 0; fn < 4; ++fn)
          p[j * N + fn * 16] = acc[fm][fn][j] + bv[fn];
    }
  }
}

// ---------------------------------------------------------------------------
// Flash attention: one WG per (b, h, 128-row q block); 8 waves x 16 q rows.
// K tile async-staged to LDS once per WG (was 8x redundant global loads);
// softmax denominator accumulated by WMMA against an all-ones B fragment.
// ---------------------------------------------------------------------------
#define QT 128
#define KT 32

__global__ __launch_bounds__(256) void attn_flash(const bf16* __restrict__ qkv,
                                                  const unsigned char* __restrict__ mask,
                                                  bf16* __restrict__ ctx) {
  __shared__ __align__(16) bf16 lK [KT][HDd];     // K tile [key][d], 4 KB
  __shared__ __align__(16) bf16 lVt[HDd][KT];     // V^T [d][key], 4 KB
  __shared__ __align__(16) bf16 lP[8][16][KT];    // per-wave P tiles, 8 KB

  const int tid  = threadIdx.x;
  const int lane = tid & 31;
  const int wave = tid >> 5;
  const int laneN    = lane & 15;
  const int laneHalf = lane >> 4;

  const int qTiles = Sd / QT;
  const int b  = blockIdx.x / (Hd * qTiles);
  const int h  = (blockIdx.x / qTiles) % Hd;
  const int q0 = (blockIdx.x % qTiles) * QT + wave * 16;

  constexpr int RS = 3 * Dd;                      // 3072
  const int koff = Dd + h * HDd;
  const int voff = 2 * Dd + h * HDd;

  // --- Q fragments (A layout) from global, pre-scaled by 1/sqrt(64)=0.125
  FragAB qf[2];
  {
    const bf16* qb = qkv + (size_t)(b * Sd + q0 + laneN) * RS + h * HDd;
#pragma unroll
    for (int ks = 0; ks < 2; ++ks) {
      const int d0 = ks * 32 + laneHalf * 8;
      qf[ks].q[0] = *(const uint4*)(qb + d0);
      qf[ks].q[1] = *(const uint4*)(qb + d0 + 16);
#pragma unroll
      for (int i = 0; i < 16; ++i)                // exact: power-of-two scale
        qf[ks].v[i] = (bf16)((float)qf[ks].v[i] * 0.125f);
    }
  }

  FragAB onesf;                                   // all-ones B frag (row sums)
#pragma unroll
  for (int i = 0; i < 16; ++i) onesf.v[i] = (bf16)1.0f;

  // --- hoisted base pointers
  const unsigned char* mbase =
      mask + (size_t)b * Sd * Sd + (size_t)(q0 + laneHalf * 8) * Sd + laneN;

  // staging map: thread -> (key, 8-dim chunk); same for K (async) and V (reg)
  const int skey = tid >> 3;
  const int sdc  = (tid & 7) * 8;
  const bf16* kg = qkv + (size_t)(b * Sd + skey) * RS + koff + sdc;
  const bf16* vp = qkv + (size_t)(b * Sd + skey) * RS + voff + sdc;
  uint4 vreg = *(const uint4*)vp;

  // per-lane LDS read base for score B frags (immediate offsets in loop)
  const bf16* lkb = &lK[laneN][laneHalf * 8];

  float mcur[8];
  v8f acc[4], lsum;
#pragma unroll
  for (int j = 0; j < 8; ++j) mcur[j] = -3.0e38f;
#pragma unroll
  for (int f = 0; f < 4; ++f) acc[f] = (v8f){0,0,0,0,0,0,0,0};
  lsum = (v8f){0,0,0,0,0,0,0,0};

  for (int kb = 0; kb < Sd; kb += KT) {
    __syncthreads();                               // prev-iter readers done
    cp16_g2l(&lK[skey][sdc], kg + (size_t)kb * RS);  // async K tile
    {
      const bf16* e = (const bf16*)&vreg;            // V transpose (overlaps)
#pragma unroll
      for (int j = 0; j < 8; ++j) lVt[sdc + j][skey] = e[j];
    }
    wait_async_0();
    __syncthreads();
    if (kb + KT < Sd) vreg = *(const uint4*)(vp + (size_t)(kb + KT) * RS);

    // --- S = Q K^T (16 rows x 32 keys); Q pre-scaled; K from LDS
    const unsigned char* mb = mbase + kb;
    float sv[2][8];
#pragma unroll
    for (int fn = 0; fn < 2; ++fn) {
      v8f s = (v8f){0,0,0,0,0,0,0,0};
#pragma unroll
      for (int ks = 0; ks < 2; ++ks) {
        FragAB kf;                                 // B layout: lane=key, table-order d
        kf.q[0] = *(const uint4*)(lkb + fn * 16 * HDd + ks * 32);
        kf.q[1] = *(const uint4*)(lkb + fn * 16 * HDd + ks * 32 + 16);
        s = WMMA_BF16(qf[ks].v, kf.v, s);
      }
#pragma unroll
      for (int j = 0; j < 8; ++j) {
        unsigned char m8 = mb[j * Sd + fn * 16];
        sv[fn][j] = m8 ? s[j] : -50000.0f;
      }
    }

    // --- online softmax: permlane16 max butterfly, WMMA row sums
    float pf[2][8];
#pragma unroll
    for (int j = 0; j < 8; ++j) {
      float v = rowmax16(fmaxf(sv[0][j], sv[1][j]));
      float mnew = fmaxf(mcur[j], v);
      float corr = __expf(mcur[j] - mnew);
      pf[0][j] = __expf(sv[0][j] - mnew);
      pf[1][j] = __expf(sv[1][j] - mnew);
      mcur[j] = mnew;
      lsum[j] *= corr;
#pragma unroll
      for (int f = 0; f < 4; ++f) acc[f][j] *= corr;
    }

    // --- P: C layout -> A layout via wave-private LDS
#pragma unroll
    for (int fn = 0; fn < 2; ++fn)
#pragma unroll
      for (int j = 0; j < 8; ++j)
        lP[wave][j + laneHalf * 8][fn * 16 + laneN] = (bf16)pf[fn][j];
    asm volatile("s_wait_dscnt 0" ::: "memory");

    FragAB pfrag;
    {
      const int c = laneHalf * 8;
      pfrag.q[0] = *(uint4*)&lP[wave][laneN][c];
      pfrag.q[1] = *(uint4*)&lP[wave][laneN][c + 16];
    }

    // --- O += P V ; lsum += P 1  (denominator via matrix unit)
    lsum = WMMA_BF16(pfrag.v, onesf.v, lsum);
#pragma unroll
    for (int f = 0; f < 4; ++f) {
      FragAB vf;                                   // B layout from lVt[d][k]
      const int d  = f * 16 + laneN;
      const int kk = laneHalf * 8;
      vf.q[0] = *(uint4*)&lVt[d][kk];
      vf.q[1] = *(uint4*)&lVt[d][kk + 16];
      acc[f] = WMMA_BF16(pfrag.v, vf.v, acc[f]);
    }
  }

  // --- normalize and store ctx (immediate offsets from one base)
  bf16* ctxb = ctx + (size_t)(b * Sd + q0 + laneHalf * 8) * Dd + h * HDd + laneN;
#pragma unroll
  for (int j = 0; j < 8; ++j) {
    const float inv = 1.0f / lsum[j];
#pragma unroll
    for (int f = 0; f < 4; ++f)
      ctxb[j * Dd + f * 16] = (bf16)(acc[f][j] * inv);
  }
}

// ---------------------------------------------------------------------------
extern "C" void kernel_launch(void* const* d_in, const int* in_sizes, int n_in,
                              void* d_out, int out_size, void* d_ws, size_t ws_size,
                              hipStream_t stream) {
  const float* hidden       = (const float*)d_in[0];
  const unsigned char* mask = (const unsigned char*)d_in[1];
  const float* Wqkv         = (const float*)d_in[2];
  const float* bqkv         = (const float*)d_in[3];
  const float* Wd           = (const float*)d_in[4];
  const float* bd           = (const float*)d_in[5];
  float* out                = (float*)d_out;

  const size_t MS = (size_t)Bd * Sd;                    // 4096
  bf16* hid_bf  = (bf16*)d_ws;
  bf16* wqkvT   = hid_bf + MS * Dd;                     // [3072][1024]
  bf16* wdT     = wqkvT  + (size_t)3 * Dd * Dd;         // [1024][1024]
  bf16* qkv_bf  = wdT    + (size_t)Dd * Dd;             // [4096][3072]
  bf16* ctx_bf  = qkv_bf + MS * 3 * Dd;                 // [4096][1024]

  // 1) conversions (+ weight transposes so GEMM B-tiles stage without transpose)
  {
    int n4 = (int)(MS * Dd) / 4;
    cvt_f32_bf16<<<(n4 + 255) / 256, 256, 0, stream>>>(hidden, hid_bf, n4);
  }
  cvt_transpose<<<dim3(Dd / 32, 3 * Dd / 32), 256, 0, stream>>>(Wqkv, wqkvT, Dd, 3 * Dd);
  cvt_transpose<<<dim3(Dd / 32, Dd / 32),     256, 0, stream>>>(Wd,   wdT,   Dd, Dd);

  // 2) fused QKV projection -> bf16 [4096,3072]
  gemm_bf16<true, 3072, 1024><<<(4096 / GM_TM) * (3072 / GM_TN), 256, 0, stream>>>(
      hid_bf, wqkvT, bqkv, qkv_bf, 4096);

  // 3) flash attention -> ctx bf16 [4096,1024]
  attn_flash<<<Bd * Hd * (Sd / QT), 256, 0, stream>>>(qkv_bf, mask, ctx_bf);

  // 4) dense projection -> fp32 d_out
  gemm_bf16<false, 1024, 1024><<<(4096 / GM_TM) * (1024 / GM_TN), 256, 0, stream>>>(
      ctx_bf, wdT, bd, out, 4096);
}
